// Qwen3NextGatedDeltaNetForEngine_32392643347142
// MI455X (gfx1250) — matrix operand order repaired and checked by reading, not compiled, not run on hardware
//
#include <hip/hip_runtime.h>
#include <stdint.h>

typedef __attribute__((ext_vector_type(16))) __bf16 v16bf;
typedef __attribute__((ext_vector_type(8)))  __bf16 v8bf;
typedef __attribute__((ext_vector_type(8)))  float  v8f;
typedef __attribute__((ext_vector_type(4)))  unsigned int u32x4;
typedef __attribute__((ext_vector_type(8)))  int    i32x8;
typedef __attribute__((ext_vector_type(4)))  int    i32x4;

#define S_LEN  256
#define HID    2048
#define KEYD   512
#define VALD   4096
#define QKVZ_N 9216
#define CONVD  5120
#define NHEAD  32
#define DKH    128
#define DVH    128

// ---------------------------------------------------------------------------
// TDM: async 2D bf16 tile load global -> LDS (D# built per ISA 08_async_tensor)
// ---------------------------------------------------------------------------
__device__ inline void tdm_load_2d_b16(unsigned lds_off, const void* gsrc,
                                       unsigned tile_x, unsigned tile_y,
                                       unsigned stride_x_elems) {
  union { unsigned long long q[2]; u32x4 v; } g0;
  union { unsigned long long q[4]; i32x8 v; } g1;
  unsigned long long ga = (unsigned long long)(uintptr_t)gsrc;
  // group0: count=1 | lds_addr<<32 ; global_addr | type(2)<<62
  g0.q[0] = 1ull | ((unsigned long long)lds_off << 32);
  g0.q[1] = (ga & 0x01FFFFFFFFFFFFFFull) | (2ull << 62);
  // group1: data_size=1(2B), tensor_dim0/1 = tile dims (tile always in-bounds),
  // tile_dim0/1, tensor_dim0_stride = row stride in elements
  unsigned long long td0 = tile_x, td1 = tile_y, st0 = stride_x_elems;
  g1.q[0] = (1ull << 16) | (td0 << 48);
  g1.q[1] = (td0 >> 16) | (td1 << 16) | ((unsigned long long)tile_x << 48);
  g1.q[2] = (unsigned long long)tile_y | (st0 << 32);
  g1.q[3] = (st0 >> 32);
  i32x4 z4 = {0, 0, 0, 0};
#if __clang_major__ >= 23
  i32x8 z8 = {0, 0, 0, 0, 0, 0, 0, 0};
  __builtin_amdgcn_tensor_load_to_lds(g0.v, g1.v, z4, z4, z8, 0);
#else
  __builtin_amdgcn_tensor_load_to_lds(g0.v, g1.v, z4, z4, 0);
#endif
}

// ---------------------------------------------------------------------------
// WMMA fragment helpers (wave32, 16x16x32 bf16)
// A frag, lane(r=lane&15,hi=lane>>4): row=r, k = {hi*8..hi*8+7, 16+hi*8..+7}
// B frag: col=r, k = {hi*8..hi*8+7, 16+hi*8..+7} (same order as A)
// ---------------------------------------------------------------------------
__device__ inline v16bf frag_a16(const __bf16* rowbase, int hi) {
  v8bf lo = *(const v8bf*)(rowbase + hi * 8);
  v8bf hv = *(const v8bf*)(rowbase + 16 + hi * 8);
  return __builtin_shufflevector(lo, hv, 0,1,2,3,4,5,6,7,8,9,10,11,12,13,14,15);
}
__device__ inline v16bf frag_b16(const __bf16* colbase, int hi) {
  v8bf lo = *(const v8bf*)(colbase + hi * 8);
  v8bf hv = *(const v8bf*)(colbase + 16 + hi * 8);
  return __builtin_shufflevector(lo, hv, 0,1,2,3,4,5,6,7,8,9,10,11,12,13,14,15);
}
__device__ inline v8f wmma_bf16(v16bf a, v16bf b, v8f c) {
  return __builtin_amdgcn_wmma_f32_16x16x32_bf16(false, a, false, b, (short)0, c,
                                                 false, false);
}
__device__ inline v8f vzero8() {
  v8f z = {0.f, 0.f, 0.f, 0.f, 0.f, 0.f, 0.f, 0.f};
  return z;
}
__device__ inline float sigmoidf_(float x) { return 1.f / (1.f + __expf(-x)); }
__device__ inline float siluf_(float x) { return x * sigmoidf_(x); }

// ---------------------------------------------------------------------------
// fp32 -> bf16 conversion (streaming)
// ---------------------------------------------------------------------------
__global__ void cvt_f32_bf16(const float* __restrict__ src,
                             __bf16* __restrict__ dst, long long n) {
  long long i = (long long)blockIdx.x * blockDim.x + threadIdx.x;
  if (i < n) dst[i] = (__bf16)src[i];
}

// ---------------------------------------------------------------------------
// Y[M,N] = Xb[M,K] @ Wb[N,K]^T  (fp32 out). Block tile 128x128, 8 waves,
// TDM double-buffered LDS staging, v_wmma_f32_16x16x32_bf16.
// M,N multiples of 128; K multiple of 32.
// ---------------------------------------------------------------------------
__global__ __launch_bounds__(256, 1)
void gemm_xwT_bf16(const __bf16* __restrict__ X, const __bf16* __restrict__ W,
                   float* __restrict__ Y, int M, int N, int K) {
  __shared__ __bf16 lA[2][128 * 32];
  __shared__ __bf16 lB[2][128 * 32];
  const int tid = threadIdx.x, wid = tid >> 5, lane = tid & 31;
  const int r = lane & 15, hi = lane >> 4;
  const int m0 = blockIdx.y * 128, n0 = blockIdx.x * 128;
  const int wm = (wid & 1) * 64;    // 2 waves along M
  const int wn = (wid >> 1) * 32;   // 4 waves along N
  const int nk = K >> 5;

  if (tid == 0) {
    tdm_load_2d_b16((unsigned)(uintptr_t)&lA[0][0], X + (size_t)m0 * K, 32, 128,
                    (unsigned)K);
    tdm_load_2d_b16((unsigned)(uintptr_t)&lB[0][0], W + (size_t)n0 * K, 32, 128,
                    (unsigned)K);
  }
  v8f acc[4][2];
#pragma unroll
  for (int i = 0; i < 4; ++i)
#pragma unroll
    for (int j = 0; j < 2; ++j) acc[i][j] = vzero8();

  for (int ks = 0; ks < nk; ++ks) {
    const int cur = ks & 1;
    if (tid == 0) {
      if (ks + 1 < nk) {
        tdm_load_2d_b16((unsigned)(uintptr_t)&lA[cur ^ 1][0],
                        X + (size_t)m0 * K + (size_t)(ks + 1) * 32, 32, 128,
                        (unsigned)K);
        tdm_load_2d_b16((unsigned)(uintptr_t)&lB[cur ^ 1][0],
                        W + (size_t)n0 * K + (size_t)(ks + 1) * 32, 32, 128,
                        (unsigned)K);
        __builtin_amdgcn_s_wait_tensorcnt(2);  // current tile's 2 ops done
      } else {
        __builtin_amdgcn_s_wait_tensorcnt(0);
      }
    }
    __syncthreads();
    v16bf bf0 = frag_b16(&lB[cur][(wn + r) * 32], hi);
    v16bf bf1 = frag_b16(&lB[cur][(wn + 16 + r) * 32], hi);
#pragma unroll
    for (int i = 0; i < 4; ++i) {
      v16bf af = frag_a16(&lA[cur][(wm + i * 16 + r) * 32], hi);
      acc[i][0] = wmma_bf16(af, bf0, acc[i][0]);
      acc[i][1] = wmma_bf16(af, bf1, acc[i][1]);
    }
    __syncthreads();
  }
#pragma unroll
  for (int i = 0; i < 4; ++i)
#pragma unroll
    for (int j = 0; j < 2; ++j) {
      const int col = n0 + wn + j * 16 + r;
#pragma unroll
      for (int e = 0; e < 8; ++e) {
        const int row = m0 + wm + i * 16 + hi * 8 + e;
        Y[(size_t)row * N + col] = acc[i][j][e];
      }
    }
}

// ---------------------------------------------------------------------------
// ba = hs @ w_ba^T   [256,64]
// ---------------------------------------------------------------------------
__global__ void ba_proj(const float* __restrict__ hs,
                        const float* __restrict__ w_ba,
                        float* __restrict__ ba) {
  const int s = blockIdx.x, j = threadIdx.x;  // 64 threads
  const float* x = hs + (size_t)s * HID;
  const float* w = w_ba + (size_t)j * HID;
  float acc = 0.f;
  for (int k = 0; k < HID; k += 4) {
    acc += x[k] * w[k] + x[k + 1] * w[k + 1] + x[k + 2] * w[k + 2] +
           x[k + 3] * w[k + 3];
  }
  ba[s * 64 + j] = acc;
}

// ---------------------------------------------------------------------------
// Causal depthwise conv (K=4) + silu over qkvz[:, :5120]; emits bf16:
//  q * DK^-0.5, k, v * sigmoid(beta + dt_bias)
// ---------------------------------------------------------------------------
__global__ void conv_silu(const float* __restrict__ qkvz,
                          const float* __restrict__ ba,
                          const float* __restrict__ conv_w,
                          const float* __restrict__ dt_bias,
                          __bf16* __restrict__ qb, __bf16* __restrict__ kb,
                          __bf16* __restrict__ vb) {
  const int c = blockIdx.x * blockDim.x + threadIdx.x;  // 5120 channels
  const int s = blockIdx.y;
  if (c >= CONVD) return;
  float acc = 0.f;
#pragma unroll
  for (int j = 0; j < 4; ++j) {
    const int t = s - 3 + j;
    if (t >= 0) acc += qkvz[(size_t)t * QKVZ_N + c] * conv_w[c * 4 + j];
  }
  const float a = siluf_(acc);
  if (c < KEYD) {
    qb[(size_t)s * KEYD + c] = (__bf16)(a * 0.08838834764831845f);  // DK^-0.5
  } else if (c < 2 * KEYD) {
    kb[(size_t)s * KEYD + (c - KEYD)] = (__bf16)a;
  } else {
    const int cc = c - 2 * KEYD;
    const int n = cc >> 7;
    const float bg = sigmoidf_(ba[s * 64 + n] + dt_bias[n]);
    vb[(size_t)s * VALD + cc] = (__bf16)(a * bg);
  }
}

// ---------------------------------------------------------------------------
// Per-head causal linear attention: O[s,:] = sum_{t<=s} (q_s . k_t) * vb[t,:]
// One block per head; 8 waves x 32 q-rows. WMMA QK^T -> mask -> LDS relayout
// (C-layout -> A-layout) -> WMMA score*V. 32-row t-tiles staged in LDS.
// ---------------------------------------------------------------------------
__global__ __launch_bounds__(256, 1)
void attn_head(const __bf16* __restrict__ Qb, const __bf16* __restrict__ Kb,
               const __bf16* __restrict__ Vb, float* __restrict__ O) {
  const int h = blockIdx.x, qh = h >> 3;  // GQA: 32 v-heads share 4 qk-heads
  const int tid = threadIdx.x, wid = tid >> 5, lane = tid & 31;
  const int r = lane & 15, hi = lane >> 4;
  const int qrow0 = wid * 32;

  __shared__ __bf16 lK[32 * 128];        // [t][dk]
  __shared__ __bf16 lVt[128 * 32];       // transposed [dv][t]
  __shared__ __bf16 lS[8][16 * 32];      // per-wave score relayout scratch

  // preload Q fragments for this wave's 32 rows
  v16bf qf[2][4];
#pragma unroll
  for (int m = 0; m < 2; ++m)
#pragma unroll
    for (int ks = 0; ks < 4; ++ks)
      qf[m][ks] = frag_a16(
          Qb + (size_t)(qrow0 + m * 16 + r) * KEYD + qh * DKH + ks * 32, hi);

  v8f oacc[2][8];
#pragma unroll
  for (int m = 0; m < 2; ++m)
#pragma unroll
    for (int d = 0; d < 8; ++d) oacc[m][d] = vzero8();

  for (int t0 = 0; t0 < S_LEN; t0 += 32) {
    // stage K tile (vectorized) and V tile (transposed)
    for (int c = tid; c < 32 * 16; c += 256) {
      const int t = c >> 4, cc = (c & 15) * 8;
      *(v8bf*)&lK[t * 128 + cc] =
          *(const v8bf*)(Kb + (size_t)(t0 + t) * KEYD + qh * DKH + cc);
    }
    {
      const int t = tid >> 3, d0 = (tid & 7) * 16;
#pragma unroll
      for (int d = 0; d < 16; ++d)
        lVt[(d0 + d) * 32 + t] =
            Vb[(size_t)(t0 + t) * VALD + h * DVH + d0 + d];
    }
    __syncthreads();

    if (t0 <= qrow0 + 31) {  // wave-uniform causal skip
#pragma unroll
      for (int m = 0; m < 2; ++m) {
        v8f sc0 = vzero8(), sc1 = vzero8();
#pragma unroll
        for (int ks = 0; ks < 4; ++ks) {
          v16bf b0 = frag_b16(&lK[r * 128 + ks * 32], hi);
          v16bf b1 = frag_b16(&lK[(16 + r) * 128 + ks * 32], hi);
          sc0 = wmma_bf16(qf[m][ks], b0, sc0);
          sc1 = wmma_bf16(qf[m][ks], b1, sc1);
        }
        // causal mask, bf16, write in C-layout; reread in A-layout
        __bf16* sp = &lS[wid][0];
#pragma unroll
        for (int e = 0; e < 8; ++e) {
          const int lr = hi * 8 + e;
          const int sg = qrow0 + m * 16 + lr;
          sp[lr * 32 + r]      = (__bf16)((t0 + r      <= sg) ? sc0[e] : 0.f);
          sp[lr * 32 + 16 + r] = (__bf16)((t0 + 16 + r <= sg) ? sc1[e] : 0.f);
        }
        asm volatile("s_wait_dscnt 0" ::: "memory");  // wave-local LDS RAW
        v16bf sfrag = frag_a16(&sp[r * 32], hi);
#pragma unroll
        for (int dv = 0; dv < 8; ++dv) {
          v16bf vf = frag_b16(&lVt[(dv * 16 + r) * 32], hi);
          oacc[m][dv] = wmma_bf16(sfrag, vf, oacc[m][dv]);
        }
      }
    }
    __syncthreads();
  }
#pragma unroll
  for (int m = 0; m < 2; ++m)
#pragma unroll
    for (int dv = 0; dv < 8; ++dv)
#pragma unroll
      for (int e = 0; e < 8; ++e) {
        const int srow = qrow0 + m * 16 + hi * 8 + e;
        O[(size_t)srow * VALD + h * DVH + dv * 16 + r] = oacc[m][dv][e];
      }
}

// ---------------------------------------------------------------------------
// Gated RMSNorm over DV then * silu(z); emits bf16 for final GEMM
// ---------------------------------------------------------------------------
__global__ void rms_gate(const float* __restrict__ O,
                         const float* __restrict__ qkvz,
                         const float* __restrict__ norm_w,
                         __bf16* __restrict__ G) {
  const int s = blockIdx.x, n = blockIdx.y, d = threadIdx.x;  // 128 threads
  const float o = O[(size_t)s * VALD + n * DVH + d];
  float ss = o * o;
#pragma unroll
  for (int off = 16; off; off >>= 1) ss += __shfl_xor(ss, off, 32);
  __shared__ float part[4];
  if ((threadIdx.x & 31) == 0) part[threadIdx.x >> 5] = ss;
  __syncthreads();
  const float var = (part[0] + part[1] + part[2] + part[3]) * (1.f / 128.f);
  const float y = o * rsqrtf(var + 1e-6f) * norm_w[d];
  const float z = qkvz[(size_t)s * QKVZ_N + CONVD + n * DVH + d];
  G[(size_t)s * VALD + n * DVH + d] = (__bf16)(y * siluf_(z));
}

// ---------------------------------------------------------------------------
extern "C" void kernel_launch(void* const* d_in, const int* in_sizes, int n_in,
                              void* d_out, int out_size, void* d_ws,
                              size_t ws_size, hipStream_t stream) {
  (void)in_sizes; (void)n_in; (void)out_size; (void)ws_size;
  const float* hs      = (const float*)d_in[0];
  const float* w_qkvz  = (const float*)d_in[1];
  const float* w_ba    = (const float*)d_in[2];
  const float* w_out   = (const float*)d_in[3];
  const float* conv_w  = (const float*)d_in[4];
  const float* dt_bias = (const float*)d_in[5];
  // d_in[6] = A_log: dead code in reference
  const float* norm_w  = (const float*)d_in[7];

  char* ws = (char*)d_ws;
  size_t off = 0;
  auto take = [&](size_t bytes) -> void* {
    void* p = ws + off;
    off += (bytes + 255) & ~(size_t)255;
    return p;
  };
  __bf16* Xb   = (__bf16*)take((size_t)S_LEN * HID * 2);
  __bf16* Wqb  = (__bf16*)take((size_t)QKVZ_N * HID * 2);
  __bf16* Wob  = (__bf16*)take((size_t)HID * VALD * 2);
  float*  qkvz = (float*) take((size_t)S_LEN * QKVZ_N * 4);
  float*  ba   = (float*) take((size_t)S_LEN * 64 * 4);
  __bf16* Qb   = (__bf16*)take((size_t)S_LEN * KEYD * 2);
  __bf16* Kb   = (__bf16*)take((size_t)S_LEN * KEYD * 2);
  __bf16* Vbb  = (__bf16*)take((size_t)S_LEN * VALD * 2);
  float*  Oat  = (float*) take((size_t)S_LEN * VALD * 4);
  __bf16* Gb   = (__bf16*)take((size_t)S_LEN * VALD * 2);

  const long long nx = (long long)S_LEN * HID;
  const long long nq = (long long)QKVZ_N * HID;
  const long long no = (long long)HID * VALD;
  cvt_f32_bf16<<<(unsigned)((nx + 255) / 256), 256, 0, stream>>>(hs, Xb, nx);
  cvt_f32_bf16<<<(unsigned)((nq + 255) / 256), 256, 0, stream>>>(w_qkvz, Wqb, nq);
  cvt_f32_bf16<<<(unsigned)((no + 255) / 256), 256, 0, stream>>>(w_out, Wob, no);

  gemm_xwT_bf16<<<dim3(QKVZ_N / 128, S_LEN / 128), 256, 0, stream>>>(
      Xb, Wqb, qkvz, S_LEN, QKVZ_N, HID);
  ba_proj<<<S_LEN, 64, 0, stream>>>(hs, w_ba, ba);
  conv_silu<<<dim3(CONVD / 256, S_LEN), 256, 0, stream>>>(qkvz, ba, conv_w,
                                                          dt_bias, Qb, Kb, Vbb);
  attn_head<<<NHEAD, 256, 0, stream>>>(Qb, Kb, Vbb, Oat);
  rms_gate<<<dim3(S_LEN, NHEAD), 128, 0, stream>>>(Oat, qkvz, norm_w, Gb);
  gemm_xwT_bf16<<<dim3(HID / 128, S_LEN / 128), 256, 0, stream>>>(
      Gb, Wob, (float*)d_out, S_LEN, HID, VALD);
}
